// FastEGNNLayer_20426864459924
// MI455X (gfx1250) — matrix operand order repaired and checked by reading, not compiled, not run on hardware
//
#include <hip/hip_runtime.h>
#include <hip/hip_bf16.h>

#define H 128
#define NNODES 50000
#define NEDGES 800000

typedef __attribute__((ext_vector_type(16))) __bf16        v16bf;
typedef __attribute__((ext_vector_type(8)))  float         v8f;
typedef __attribute__((ext_vector_type(8)))  unsigned int  v8u;

// ---------- helpers ----------

// Pack two f32 into a bf16 pair: mantissa-round then splice high halves with
// one v_perm_b32.  3 VALU ops per pair (vs ~7 + b16-mov storm for exact RNE).
__device__ __forceinline__ unsigned int pk_bf16(float lo, float hi) {
    const unsigned int a = __float_as_uint(lo) + 0x8000u;
    const unsigned int b = __float_as_uint(hi) + 0x8000u;
    // result bytes: [1:0] = a[3:2], [3:2] = b[3:2]  (S0=b -> idx 4..7, S1=a -> idx 0..3)
    return __builtin_amdgcn_perm(b, a, 0x07060302u);
}

__device__ __forceinline__ v8f vzero8() {
    v8f r;
#pragma unroll
    for (int i = 0; i < 8; ++i) r[i] = 0.0f;
    return r;
}

__device__ __forceinline__ v8f wmma_bf16(v8u a, v8u b, v8f c) {
    return __builtin_amdgcn_wmma_f32_16x16x32_bf16(
        false, __builtin_bit_cast(v16bf, a),
        false, __builtin_bit_cast(v16bf, b),
        (short)0, c, false, false);
}

// Build one 16x32 bf16 A-operand for this lane from a row of f32 data.
// p must point at  row_base + kt*32 + half*8.
// ISA A layout: lane(m=L%16, half=L/16); VGPR v, elem e ->
//   K = (v<4 ? 0 : 16) + half*8 + (v%4)*2 + e  (two contiguous 8-float runs)
__device__ __forceinline__ v8u packA(const float* p) {
    const float4 f0 = *(const float4*)(p);
    const float4 f1 = *(const float4*)(p + 4);
    const float4 f2 = *(const float4*)(p + 16);
    const float4 f3 = *(const float4*)(p + 20);
    v8u a;
    a[0] = pk_bf16(f0.x, f0.y);  a[1] = pk_bf16(f0.z, f0.w);
    a[2] = pk_bf16(f1.x, f1.y);  a[3] = pk_bf16(f1.z, f1.w);
    a[4] = pk_bf16(f2.x, f2.y);  a[5] = pk_bf16(f2.z, f2.w);
    a[6] = pk_bf16(f3.x, f3.y);  a[7] = pk_bf16(f3.z, f3.w);
    return a;
}

__device__ __forceinline__ float sigmoid_f(float v) { return 1.0f / (1.0f + __expf(-v)); }
__device__ __forceinline__ float silu_f(float v)    { return v * sigmoid_f(v); }

// sum over the 16 lanes of this lane's half-wave (keeps bit4 intact)
__device__ __forceinline__ float hsum16(float p) {
    p += __shfl_xor(p, 1, 32);
    p += __shfl_xor(p, 2, 32);
    p += __shfl_xor(p, 4, 32);
    p += __shfl_xor(p, 8, 32);
    return p;
}

// ---------- weight pre-swizzle: f32 (K,128) row-major -> bf16 B-tiles ----------
// Tile t = kt*8+nt holds a 32x16 -> 16x16 B operand: dst[t*256 + lane*8 + v]
// packs W[k][n], W[k+1][n] with k = kt*32 + (v/4)*16 + (lane/16)*8 + (v%4)*2, n = nt*16 + lane%16.
__global__ void egnn_prep_kernel(const float* __restrict__ W,
                                 unsigned int* __restrict__ dst, int nTiles) {
    int tid = blockIdx.x * blockDim.x + threadIdx.x;
    if (tid >= nTiles * 256) return;
    const int tile = tid >> 8;
    const int rem  = tid & 255;
    const int lane = rem >> 3;
    const int v    = rem & 7;
    const int kt = tile >> 3, nt = tile & 7;
    const int nn = lane & 15, hb = lane >> 4;
    const int k   = kt * 32 + (v >> 2) * 16 + hb * 8 + (v & 3) * 2;
    const int col = nt * 16 + nn;
    dst[tid] = pk_bf16(W[k * H + col], W[(k + 1) * H + col]);
}

__global__ void egnn_zero_kernel(float* __restrict__ p, int count) {
    int i = blockIdx.x * blockDim.x + threadIdx.x;
    if (i < count) p[i] = 0.0f;
}

// ---------- edge kernel: one 16-edge tile per wave, 8 waves/block ----------
__global__ __launch_bounds__(256) void egnn_edge_kernel(
    const float* __restrict__ h, const float* __restrict__ x,
    const long long* __restrict__ ei, const float* __restrict__ dist,
    const float* __restrict__ We1, const float* __restrict__ be1,
    const float* __restrict__ be2,
    const float* __restrict__ Wa, const float* __restrict__ ba,
    const float* __restrict__ bc1, const float* __restrict__ Wc2,
    const v8u* __restrict__ We1s, const v8u* __restrict__ We2s,
    const v8u* __restrict__ Wc1s,
    float* __restrict__ agg_msg, float* __restrict__ agg_coord) {

    __shared__ float lds[8][16 * H];   // 64 KB: per-wave 16x128 f32 transpose buffer
    const int lane = threadIdx.x & 31;
    const int wave = threadIdx.x >> 5;
    const int tile = blockIdx.x * 8 + wave;     // E/128 = 6250 blocks exactly
    const int e0   = tile * 16;
    const int n    = lane & 15;
    const int half = lane >> 4;
    float* lw = lds[wave];

    // pull L2-resident swizzled weights toward the WGP before the WMMA burst
    __builtin_prefetch(&We1s[lane], 0, 3);
    __builtin_prefetch(&We2s[lane], 0, 3);
    __builtin_prefetch(&Wc1s[lane], 0, 3);

    // A-operand gather rows: this lane supplies edge m = n
    const long long rowm = ei[e0 + n];
    const long long colm = ei[NEDGES + e0 + n];
    const float* hr = h + rowm * H;
    const float* hc = h + colm * H;

    // ---- t1 = [h_row | h_col | dist] @ We1 + be1 ----
    v8f acc[8];
#pragma unroll
    for (int nt = 0; nt < 8; ++nt) acc[nt] = vzero8();

#pragma unroll
    for (int kt = 0; kt < 4; ++kt) {
        const v8u a = packA(hr + kt * 32 + half * 8);
#pragma unroll
        for (int nt = 0; nt < 8; ++nt)
            acc[nt] = wmma_bf16(a, We1s[(kt * 8 + nt) * 32 + lane], acc[nt]);
    }
#pragma unroll
    for (int kt = 0; kt < 4; ++kt) {
        const v8u a = packA(hc + kt * 32 + half * 8);
#pragma unroll
        for (int nt = 0; nt < 8; ++nt)
            acc[nt] = wmma_bf16(a, We1s[((kt + 4) * 8 + nt) * 32 + lane], acc[nt]);
    }

    // dist column (row 256 of We1) as exact f32 rank-1 update, + bias, + SiLU
    const float4 d0 = *(const float4*)(dist + e0 + half * 8);
    const float4 d1 = *(const float4*)(dist + e0 + half * 8 + 4);
    const float dv[8] = {d0.x, d0.y, d0.z, d0.w, d1.x, d1.y, d1.z, d1.w};
#pragma unroll
    for (int nt = 0; nt < 8; ++nt) {
        const float w256 = We1[256 * H + nt * 16 + n];
        const float b1   = be1[nt * 16 + n];
#pragma unroll
        for (int v = 0; v < 8; ++v)
            acc[nt][v] = silu_f(acc[nt][v] + dv[v] * w256 + b1);
    }

    // D-layout -> LDS row-major [m][j]
#pragma unroll
    for (int nt = 0; nt < 8; ++nt)
#pragma unroll
        for (int v = 0; v < 8; ++v)
            lw[(v + 8 * half) * H + nt * 16 + n] = acc[nt][v];
    __syncthreads();

    // ---- t2 = silu(t1) @ We2 + be2 ----
    v8f acc2[8];
#pragma unroll
    for (int nt = 0; nt < 8; ++nt) acc2[nt] = vzero8();
#pragma unroll
    for (int kt = 0; kt < 4; ++kt) {
        const v8u a = packA(lw + n * H + kt * 32 + half * 8);
#pragma unroll
        for (int nt = 0; nt < 8; ++nt)
            acc2[nt] = wmma_bf16(a, We2s[(kt * 8 + nt) * 32 + lane], acc2[nt]);
    }
    __syncthreads();

    // ---- attention: attn = sigmoid(t2 @ Wa + ba); t2 *= attn ----
    float pa[8] = {0, 0, 0, 0, 0, 0, 0, 0};
#pragma unroll
    for (int nt = 0; nt < 8; ++nt) {
        const float b2 = be2[nt * 16 + n];
        const float wa = Wa[nt * 16 + n];
#pragma unroll
        for (int v = 0; v < 8; ++v) {
            acc2[nt][v] += b2;
            pa[v] += acc2[nt][v] * wa;
        }
    }
    const float ba0 = ba[0];
    float at[8];
#pragma unroll
    for (int v = 0; v < 8; ++v) at[v] = sigmoid_f(hsum16(pa[v]) + ba0);
#pragma unroll
    for (int nt = 0; nt < 8; ++nt)
#pragma unroll
        for (int v = 0; v < 8; ++v) acc2[nt][v] *= at[v];

    // edge_msg -> LDS (for coord GEMM A-operand and for the scatter)
#pragma unroll
    for (int nt = 0; nt < 8; ++nt)
#pragma unroll
        for (int v = 0; v < 8; ++v)
            lw[(v + 8 * half) * H + nt * 16 + n] = acc2[nt][v];
    __syncthreads();

    // ---- coord_w = silu(edge_msg @ Wc1 + bc1) @ Wc2 ----
    v8f acc3[8];
#pragma unroll
    for (int nt = 0; nt < 8; ++nt) acc3[nt] = vzero8();
#pragma unroll
    for (int kt = 0; kt < 4; ++kt) {
        const v8u a = packA(lw + n * H + kt * 32 + half * 8);
#pragma unroll
        for (int nt = 0; nt < 8; ++nt)
            acc3[nt] = wmma_bf16(a, Wc1s[(kt * 8 + nt) * 32 + lane], acc3[nt]);
    }
    float pc[8] = {0, 0, 0, 0, 0, 0, 0, 0};
#pragma unroll
    for (int nt = 0; nt < 8; ++nt) {
        const float bc = bc1[nt * 16 + n];
        const float wc = Wc2[nt * 16 + n];
#pragma unroll
        for (int v = 0; v < 8; ++v)
            pc[v] += silu_f(acc3[nt][v] + bc) * wc;
    }
    float cw[8];
#pragma unroll
    for (int v = 0; v < 8; ++v) cw[v] = hsum16(pc[v]);

    // ---- scatter agg_msg: each lane adds 4 contiguous features per edge ----
    for (int m = 0; m < 16; ++m) {
        const long long cj = ei[NEDGES + e0 + m];
        const float4 val = *(const float4*)(lw + m * H + lane * 4);
        float* dstp = agg_msg + cj * H + lane * 4;
        atomicAdd(dstp + 0, val.x);
        atomicAdd(dstp + 1, val.y);
        atomicAdd(dstp + 2, val.z);
        atomicAdd(dstp + 3, val.w);
    }

    // ---- scatter agg_coord: one lane per half handles its edge ----
#pragma unroll
    for (int v = 0; v < 8; ++v) {
        if (n == v) {
            const int m = v + 8 * half;
            const long long e = e0 + m;
            const long long r = ei[e];
            const long long c = ei[NEDGES + e];
            const float w = cw[v];
#pragma unroll
            for (int d = 0; d < 3; ++d)
                atomicAdd(&agg_coord[c * 3 + d], (x[r * 3 + d] - x[c * 3 + d]) * w);
        }
    }
}

// ---------- node kernel: one 16-node tile per wave ----------
__global__ __launch_bounds__(256) void egnn_node_kernel(
    const float* __restrict__ h, const float* __restrict__ agg,
    const v8u* __restrict__ Wn1s, const v8u* __restrict__ Wn2s,
    const float* __restrict__ bn1, const float* __restrict__ bn2,
    const float* __restrict__ gamma, const float* __restrict__ beta,
    float* __restrict__ out_h) {

    __shared__ float lds[8][16 * H];
    const int lane = threadIdx.x & 31;
    const int wave = threadIdx.x >> 5;
    const int NT = NNODES / 16;
    int tile = blockIdx.x * 8 + wave;
    if (tile >= NT) tile = NT - 1;       // duplicate writes of identical values
    const int n0   = tile * 16;
    const int n    = lane & 15;
    const int half = lane >> 4;
    float* lw = lds[wave];

    __builtin_prefetch(&Wn1s[lane], 0, 3);
    __builtin_prefetch(&Wn2s[lane], 0, 3);

    const float* hr = h + (long long)(n0 + n) * H;
    const float* ar = agg + (long long)(n0 + n) * H;

    // ---- t = silu([h | agg] @ Wn1 + bn1) ----
    v8f acc[8];
#pragma unroll
    for (int nt = 0; nt < 8; ++nt) acc[nt] = vzero8();
#pragma unroll
    for (int kt = 0; kt < 4; ++kt) {
        const v8u a = packA(hr + kt * 32 + half * 8);
#pragma unroll
        for (int nt = 0; nt < 8; ++nt)
            acc[nt] = wmma_bf16(a, Wn1s[(kt * 8 + nt) * 32 + lane], acc[nt]);
    }
#pragma unroll
    for (int kt = 0; kt < 4; ++kt) {
        const v8u a = packA(ar + kt * 32 + half * 8);
#pragma unroll
        for (int nt = 0; nt < 8; ++nt)
            acc[nt] = wmma_bf16(a, Wn1s[((kt + 4) * 8 + nt) * 32 + lane], acc[nt]);
    }
#pragma unroll
    for (int nt = 0; nt < 8; ++nt) {
        const float b1 = bn1[nt * 16 + n];
#pragma unroll
        for (int v = 0; v < 8; ++v) acc[nt][v] = silu_f(acc[nt][v] + b1);
    }
#pragma unroll
    for (int nt = 0; nt < 8; ++nt)
#pragma unroll
        for (int v = 0; v < 8; ++v)
            lw[(v + 8 * half) * H + nt * 16 + n] = acc[nt][v];
    __syncthreads();

    // ---- h_upd = t @ Wn2 + bn2 ; z = h + h_upd ----
    v8f acc2[8];
#pragma unroll
    for (int nt = 0; nt < 8; ++nt) acc2[nt] = vzero8();
#pragma unroll
    for (int kt = 0; kt < 4; ++kt) {
        const v8u a = packA(lw + n * H + kt * 32 + half * 8);
#pragma unroll
        for (int nt = 0; nt < 8; ++nt)
            acc2[nt] = wmma_bf16(a, Wn2s[(kt * 8 + nt) * 32 + lane], acc2[nt]);
    }
#pragma unroll
    for (int nt = 0; nt < 8; ++nt) {
        const float b2 = bn2[nt * 16 + n];
#pragma unroll
        for (int v = 0; v < 8; ++v)
            acc2[nt][v] += b2 + h[(long long)(n0 + v + 8 * half) * H + nt * 16 + n];
    }

    // ---- LayerNorm over the 128 features (reduce over nt and over 16 lanes) ----
    float s[8] = {0, 0, 0, 0, 0, 0, 0, 0};
#pragma unroll
    for (int nt = 0; nt < 8; ++nt)
#pragma unroll
        for (int v = 0; v < 8; ++v) s[v] += acc2[nt][v];
    float mu[8];
#pragma unroll
    for (int v = 0; v < 8; ++v) mu[v] = hsum16(s[v]) * (1.0f / H);

    float q[8] = {0, 0, 0, 0, 0, 0, 0, 0};
#pragma unroll
    for (int nt = 0; nt < 8; ++nt)
#pragma unroll
        for (int v = 0; v < 8; ++v) {
            const float d = acc2[nt][v] - mu[v];
            q[v] += d * d;
        }
    float rstd[8];
#pragma unroll
    for (int v = 0; v < 8; ++v)
        rstd[v] = rsqrtf(hsum16(q[v]) * (1.0f / H) + 1e-5f);

#pragma unroll
    for (int nt = 0; nt < 8; ++nt) {
        const float g  = gamma[nt * 16 + n];
        const float bt = beta[nt * 16 + n];
#pragma unroll
        for (int v = 0; v < 8; ++v)
            out_h[(long long)(n0 + v + 8 * half) * H + nt * 16 + n] =
                (acc2[nt][v] - mu[v]) * rstd[v] * g + bt;
    }
}

__global__ void egnn_xout_kernel(const float* __restrict__ x,
                                 const float* __restrict__ aggc,
                                 float* __restrict__ outx, int count) {
    int i = blockIdx.x * blockDim.x + threadIdx.x;
    if (i < count) outx[i] = x[i] + aggc[i];
}

// ---------- host ----------
extern "C" void kernel_launch(void* const* d_in, const int* in_sizes, int n_in,
                              void* d_out, int out_size, void* d_ws, size_t ws_size,
                              hipStream_t stream) {
    const float* h        = (const float*)d_in[0];
    const float* x        = (const float*)d_in[1];
    const long long* ei   = (const long long*)d_in[2];
    const float* dist     = (const float*)d_in[3];
    const float* We1      = (const float*)d_in[4];
    const float* be1      = (const float*)d_in[5];
    const float* We2      = (const float*)d_in[6];
    const float* be2      = (const float*)d_in[7];
    const float* Wa       = (const float*)d_in[8];
    const float* ba       = (const float*)d_in[9];
    const float* Wn1      = (const float*)d_in[10];
    const float* bn1      = (const float*)d_in[11];
    const float* Wn2      = (const float*)d_in[12];
    const float* bn2      = (const float*)d_in[13];
    const float* Wc1      = (const float*)d_in[14];
    const float* bc1      = (const float*)d_in[15];
    const float* Wc2      = (const float*)d_in[16];
    const float* gammap   = (const float*)d_in[17];
    const float* betap    = (const float*)d_in[18];

    float* agg_msg   = (float*)d_ws;                         // N*128 f32
    float* agg_coord = agg_msg + (size_t)NNODES * H;         // N*3 f32
    unsigned int* We1s = (unsigned int*)(agg_coord + (size_t)NNODES * 3);
    unsigned int* We2s = We1s + 64 * 256;
    unsigned int* Wc1s = We2s + 32 * 256;
    unsigned int* Wn1s = Wc1s + 32 * 256;
    unsigned int* Wn2s = Wn1s + 64 * 256;

    const int zc = NNODES * H + NNODES * 3;
    egnn_zero_kernel<<<(zc + 255) / 256, 256, 0, stream>>>(agg_msg, zc);

    egnn_prep_kernel<<<(64 * 256 + 255) / 256, 256, 0, stream>>>(We1, We1s, 64);
    egnn_prep_kernel<<<(32 * 256 + 255) / 256, 256, 0, stream>>>(We2, We2s, 32);
    egnn_prep_kernel<<<(32 * 256 + 255) / 256, 256, 0, stream>>>(Wc1, Wc1s, 32);
    egnn_prep_kernel<<<(64 * 256 + 255) / 256, 256, 0, stream>>>(Wn1, Wn1s, 64);
    egnn_prep_kernel<<<(32 * 256 + 255) / 256, 256, 0, stream>>>(Wn2, Wn2s, 32);

    egnn_edge_kernel<<<NEDGES / (16 * 8), 256, 0, stream>>>(
        h, x, ei, dist, We1, be1, be2, Wa, ba, bc1, Wc2,
        (const v8u*)We1s, (const v8u*)We2s, (const v8u*)Wc1s,
        agg_msg, agg_coord);

    float* out_h = (float*)d_out;
    float* out_x = out_h + (size_t)NNODES * H;
    egnn_node_kernel<<<(NNODES / 16 + 7) / 8, 256, 0, stream>>>(
        h, agg_msg, (const v8u*)Wn1s, (const v8u*)Wn2s,
        bn1, bn2, gammap, betap, out_h);

    egnn_xout_kernel<<<(NNODES * 3 + 255) / 256, 256, 0, stream>>>(
        x, agg_coord, out_x, NNODES * 3);
}
